// MultiHeadAttention_56848187130190
// MI455X (gfx1250) — matrix operand order repaired and checked by reading, not compiled, not run on hardware
//
#include <hip/hip_runtime.h>

// Problem constants (from reference): B=4, S=2048, D=1024, H=16, DK=64
#define B_    4
#define S_    2048
#define D_    1024
#define H_    16
#define DK_   64
#define MTOT  (B_ * S_)          // 8192 rows for the projection GEMMs

// GEMM tiling
#define BM 128
#define BN 128
#define BK 32
#define NKI (D_ / BK)            // 32 K-steps
#define LDSS 40                  // padded LDS stride (elements); 80B rows -> 16B aligned frags
#define KSTR 72                  // K-tile LDS stride (elements); 144B rows, conflict-free b128 frag reads

typedef __bf16 bf16;
typedef __attribute__((ext_vector_type(16))) __bf16 v16bf;
typedef __attribute__((ext_vector_type(8)))  __bf16 v8bf;
typedef __attribute__((ext_vector_type(8)))  float  v8f;
typedef __attribute__((ext_vector_type(4)))  float  v4f;
typedef __attribute__((ext_vector_type(2)))  unsigned int v2u;

static __device__ __forceinline__ v16bf join16(v8bf lo, v8bf hi) {
  return __builtin_shufflevector(lo, hi, 0,1,2,3,4,5,6,7,8,9,10,11,12,13,14,15);
}

static __device__ __forceinline__ v8f wmma_bf16(v16bf a, v16bf b, v8f c) {
  // D = A(16x32 bf16) * B(32x16 bf16) + C(16x16 f32)
  return __builtin_amdgcn_wmma_f32_16x16x32_bf16(false, a, false, b, (short)0, c, false, false);
}

// xor-butterfly within each 16-lane row via DPP ROW_XMASK (0x160 | mask):
// pure VALU (v_mov_b32 dpp), no LDS pipe, co-executes with WMMA.
template <int MASK>
static __device__ __forceinline__ float dpp_xor16(float v) {
  int x = __builtin_bit_cast(int, v);
  x = __builtin_amdgcn_update_dpp(x, x, 0x160 | MASK, 0xF, 0xF, true);
  return __builtin_bit_cast(float, x);
}
static __device__ __forceinline__ float redmax16(float v) {
  v = fmaxf(v, dpp_xor16<1>(v));
  v = fmaxf(v, dpp_xor16<2>(v));
  v = fmaxf(v, dpp_xor16<4>(v));
  v = fmaxf(v, dpp_xor16<8>(v));
  return v;
}
static __device__ __forceinline__ float redsum16(float v) {
  v += dpp_xor16<1>(v);
  v += dpp_xor16<2>(v);
  v += dpp_xor16<4>(v);
  v += dpp_xor16<8>(v);
  return v;
}

static __device__ __forceinline__ unsigned short bfbits(float x) {
  return __builtin_bit_cast(unsigned short, (bf16)x);
}

// ---------------------------------------------------------------------------
// GEMM: out = A[MTOT x D] * W[D x D] + bias; fp32 streamed -> bf16 LDS tiles
// (double-buffered), computed with v_wmma_f32_16x16x32_bf16.
// MODE 0: out bf16, head-split  [B,H,S,DK]   (Q and K projections)
// MODE 1: out bf16, transposed  [B,H,DK,S]   (V projection -> Vt), packed stores
// MODE 2: out fp32, flat        [M, D]       (output projection)
// ---------------------------------------------------------------------------
template <int MODE>
__global__ __launch_bounds__(256)
void mha_gemm_kernel(const float* __restrict__ A, const float* __restrict__ W,
                     const float* __restrict__ bias, void* __restrict__ outv)
{
  __shared__ __align__(16) bf16 sA[2][BM * LDSS];   // [m][k]
  __shared__ __align__(16) bf16 sB[2][BN * LDSS];   // [n][k] (transposed weights)

  const int tid  = threadIdx.x;
  const int lane = tid & 31;
  const int half = lane >> 4;
  const int l16  = lane & 15;
  const int wave = tid >> 5;     // 0..7
  const int wm   = wave >> 1;    // 0..3 -> 32-row strip
  const int wn   = wave & 1;     // 0..1 -> 64-col strip
  const int mBase = blockIdx.y * BM;
  const int nBase = blockIdx.x * BN;

  v8f acc[2][4];
#pragma unroll
  for (int i = 0; i < 2; ++i)
#pragma unroll
    for (int j = 0; j < 4; ++j) acc[i][j] = v8f{};

  v4f xa[4], xb[4];   // in-flight staging registers (next K tile)

  auto loadTile = [&](int kt) {
#pragma unroll
    for (int i = 0; i < 4; ++i) {
      int idx = tid + 256 * i;          // 0..1023
      int row = idx >> 3;               // A: 0..127
      int kk  = (idx & 7) * 4;          // A: 0..28
      xa[i] = *reinterpret_cast<const v4f*>(A + (size_t)(mBase + row) * D_ + kt * BK + kk);
      int kr = idx >> 5;                // W: 0..31
      int n4 = (idx & 31) * 4;          // W: 0..124
      xb[i] = *reinterpret_cast<const v4f*>(W + (size_t)(kt * BK + kr) * D_ + nBase + n4);
    }
  };
  auto storeTile = [&](int buf) {
#pragma unroll
    for (int i = 0; i < 4; ++i) {
      int idx = tid + 256 * i;
      int row = idx >> 3;
      int kk  = (idx & 7) * 4;
      v2u u;
      u[0] = (unsigned)bfbits(xa[i][0]) | ((unsigned)bfbits(xa[i][1]) << 16);
      u[1] = (unsigned)bfbits(xa[i][2]) | ((unsigned)bfbits(xa[i][3]) << 16);
      *reinterpret_cast<v2u*>(&sA[buf][row * LDSS + kk]) = u;   // 8B-aligned packed store
      int kr = idx >> 5;
      int n4 = (idx & 31) * 4;
      sB[buf][(n4 + 0) * LDSS + kr] = (bf16)xb[i][0];
      sB[buf][(n4 + 1) * LDSS + kr] = (bf16)xb[i][1];
      sB[buf][(n4 + 2) * LDSS + kr] = (bf16)xb[i][2];
      sB[buf][(n4 + 3) * LDSS + kr] = (bf16)xb[i][3];
    }
  };

  loadTile(0);
  storeTile(0);
  __syncthreads();

  for (int kt = 0; kt < NKI; ++kt) {
    const int cur = kt & 1;

    // ---- fragments from current buffer (ds_loads issued first) ----
    v16bf af[2];
#pragma unroll
    for (int tm = 0; tm < 2; ++tm) {
      int row = wm * 32 + tm * 16 + l16;
      int ko  = half * 8;               // A layout: lanes 0-15 K{0..7,16..23}; 16-31 K{8..15,24..31}
      const bf16* p = &sA[cur][row * LDSS + ko];
      af[tm] = join16(*reinterpret_cast<const v8bf*>(p),
                      *reinterpret_cast<const v8bf*>(p + 16));
    }
    v16bf bfg[4];
#pragma unroll
    for (int tn = 0; tn < 4; ++tn) {
      int n  = wn * 64 + tn * 16 + l16;
      int ko = half * 8;                // B layout: lanes 0-15 K{0..7,16..23}; 16-31 K{8..15,24..31}
      const bf16* p = &sB[cur][n * LDSS + ko];
      bfg[tn] = join16(*reinterpret_cast<const v8bf*>(p),
                       *reinterpret_cast<const v8bf*>(p + 16));
    }

    // ---- issue next tile's global loads (overlap with WMMAs below) ----
    if (kt + 1 < NKI) loadTile(kt + 1);

    // ---- 8 WMMAs ----
#pragma unroll
    for (int tm = 0; tm < 2; ++tm)
#pragma unroll
      for (int tn = 0; tn < 4; ++tn)
        acc[tm][tn] = wmma_bf16(af[tm], bfg[tn], acc[tm][tn]);

    // ---- stage next tile into the other buffer ----
    if (kt + 1 < NKI) storeTile((kt + 1) & 1);
    __syncthreads();
  }

  // ---- epilogue: bias + layout-specific store ----
#pragma unroll
  for (int tm = 0; tm < 2; ++tm) {
#pragma unroll
    for (int tn = 0; tn < 4; ++tn) {
      int nG = nBase + wn * 64 + tn * 16 + l16;
      float bv = bias[nG];
      if constexpr (MODE == 1) {
        // lane's 8 values are 8 consecutive s positions in Vt -> one b128 store
        int m0 = mBase + wm * 32 + tm * 16 + 8 * half;
        int b = m0 >> 11, s0 = m0 & (S_ - 1);
        int h = nG >> 6,  dk = nG & (DK_ - 1);
        v8bf pk;
#pragma unroll
        for (int r = 0; r < 8; ++r) pk[r] = (bf16)(acc[tm][tn][r] + bv);
        *reinterpret_cast<v8bf*>(
            (bf16*)outv + (((size_t)(b * H_ + h) * DK_) + dk) * S_ + s0) = pk;
      } else {
#pragma unroll
        for (int r = 0; r < 8; ++r) {
          int mG = mBase + wm * 32 + tm * 16 + r + 8 * half;  // C layout: VGPR r -> row r / r+8
          float val = acc[tm][tn][r] + bv;
          if constexpr (MODE == 0) {
            int b = mG >> 11, s = mG & (S_ - 1);
            int h = nG >> 6,  dk = nG & (DK_ - 1);
            ((bf16*)outv)[(((size_t)(b * H_ + h) * S_) + s) * DK_ + dk] = (bf16)val;
          } else {
            ((float*)outv)[(size_t)mG * D_ + nG] = val;
          }
        }
      }
    }
  }
}

// ---------------------------------------------------------------------------
// Flash attention: one wave per 16 Q rows, kv blocks of 32.
// K (32x64) and V (64x32) tiles staged once per block into double-buffered LDS
// (4x less L2 traffic than per-wave loads); Qh,Kh: bf16 [B,H,S,DK];
// Vt: bf16 [B,H,DK,S]; ctx: fp32 [B,S,D] (head-merged).
// ---------------------------------------------------------------------------
__global__ __launch_bounds__(128)
void mha_attn_kernel(const bf16* __restrict__ Qh, const bf16* __restrict__ Kh,
                     const bf16* __restrict__ Vt, float* __restrict__ ctx)
{
  __shared__ __align__(16) bf16 sK[2][32 * KSTR];   // [kv][dk], 144B rows
  __shared__ __align__(16) bf16 sV[2][64 * LDSS];   // [dk][kv], 80B rows
  __shared__ __align__(16) bf16 sP[4][16 * LDSS];   // per-wave P scratch (C->A relayout)

  const int tid  = threadIdx.x;
  const int lane = tid & 31;
  const int half = lane >> 4;
  const int l16  = lane & 15;
  const int wave = tid >> 5;                 // 0..3

  const int qblocks = S_ / 64;               // 32
  const int bh = blockIdx.x / qblocks;       // 0..B*H-1
  const int q0 = (blockIdx.x % qblocks) * 64 + wave * 16;

  const size_t base = (size_t)bh * S_ * DK_;
  const bf16* Qp = Qh + base;
  const bf16* Kp = Kh + base;
  const bf16* Vp = Vt + base;                // [DK][S] within this (b,h)

  // Q fragments (16 rows x 64 dk -> two K=32 chunks), kept in registers
  v16bf aq[2];
  {
    int srow = q0 + l16;
    int ko   = half * 8;
#pragma unroll
    for (int c = 0; c < 2; ++c) {
      const bf16* p = Qp + (size_t)srow * DK_ + 32 * c + ko;
      aq[c] = join16(*reinterpret_cast<const v8bf*>(p),
                     *reinterpret_cast<const v8bf*>(p + 16));
    }
  }

  // cooperative K/V tile staging: 128 threads x 32B each per tile
  v8bf kreg[2], vreg[2];
  const int krow = tid >> 2, kseg = (tid & 3) * 16;   // K: [32 rows][64 dk]
  const int vrow = tid >> 1, vseg = (tid & 1) * 16;   // V: [64 rows][32 kv]
  auto loadKV = [&](int j) {
    const bf16* kp = Kp + (size_t)(j + krow) * DK_ + kseg;
    kreg[0] = *reinterpret_cast<const v8bf*>(kp);
    kreg[1] = *reinterpret_cast<const v8bf*>(kp + 8);
    const bf16* vp = Vp + (size_t)vrow * S_ + j + vseg;
    vreg[0] = *reinterpret_cast<const v8bf*>(vp);
    vreg[1] = *reinterpret_cast<const v8bf*>(vp + 8);
  };
  auto storeKV = [&](int buf) {
    *reinterpret_cast<v8bf*>(&sK[buf][krow * KSTR + kseg])     = kreg[0];
    *reinterpret_cast<v8bf*>(&sK[buf][krow * KSTR + kseg + 8]) = kreg[1];
    *reinterpret_cast<v8bf*>(&sV[buf][vrow * LDSS + vseg])     = vreg[0];
    *reinterpret_cast<v8bf*>(&sV[buf][vrow * LDSS + vseg + 8]) = vreg[1];
  };

  float mrow[8], lrow[8];
  v8f oacc[4];
#pragma unroll
  for (int r = 0; r < 8; ++r) { mrow[r] = -3.0e38f; lrow[r] = 0.0f; }
#pragma unroll
  for (int n = 0; n < 4; ++n) oacc[n] = v8f{};

  bf16* myP = sP[wave];
  const float scale = 0.125f;                // 1/sqrt(64)
  const int NJ = S_ / 32;                    // 64 kv steps

  loadKV(0);
  storeKV(0);
  __syncthreads();

  for (int jt = 0; jt < NJ; ++jt) {
    const int cur = jt & 1;

    // ---- K/V fragments from current LDS buffer (ds_loads issued first) ----
    v16bf bk[2][2];
#pragma unroll
    for (int t = 0; t < 2; ++t) {
#pragma unroll
      for (int c = 0; c < 2; ++c) {
        int kvrow = 16 * t + l16;            // WMMA-B column = kv position
        int dk0   = 32 * c + half * 8;
        const bf16* p = &sK[cur][kvrow * KSTR + dk0];
        bk[t][c] = join16(*reinterpret_cast<const v8bf*>(p),
                          *reinterpret_cast<const v8bf*>(p + 16));
      }
    }
    v16bf bv[4];
#pragma unroll
    for (int n = 0; n < 4; ++n) {
      int dk  = 16 * n + l16;                // WMMA-B column = dk
      int kv0 = half * 8;
      const bf16* p = &sV[cur][dk * LDSS + kv0];
      bv[n] = join16(*reinterpret_cast<const v8bf*>(p),
                     *reinterpret_cast<const v8bf*>(p + 16));
    }

    // ---- issue next tile's global loads (overlap with compute) ----
    if (jt + 1 < NJ) loadKV((jt + 1) * 32);

    // ---- scores: two 16x16 tiles over this kv block ----
    v8f sc[2] = { v8f{}, v8f{} };
#pragma unroll
    for (int t = 0; t < 2; ++t) {
      sc[t] = wmma_bf16(aq[0], bk[t][0], sc[t]);
      sc[t] = wmma_bf16(aq[1], bk[t][1], sc[t]);
    }

    // ---- online softmax over the 32 columns (DPP row_xmask butterflies) ----
    float e0[8], e1[8];
#pragma unroll
    for (int r = 0; r < 8; ++r) {
      float a  = sc[0][r] * scale;
      float b2 = sc[1][r] * scale;
      float v  = redmax16(fmaxf(a, b2));
      float mn = fmaxf(mrow[r], v);
      float alpha = __expf(mrow[r] - mn);
      e0[r] = __expf(a - mn);
      e1[r] = __expf(b2 - mn);
      float srt = redsum16(e0[r] + e1[r]);
      lrow[r] = lrow[r] * alpha + srt;
      mrow[r] = mn;
#pragma unroll
      for (int n = 0; n < 4; ++n) oacc[n][r] *= alpha;
    }

    // ---- P: C-layout -> bf16 A-layout via per-wave LDS ----
#pragma unroll
    for (int r = 0; r < 8; ++r) {
      int row = r + 8 * half;
      myP[row * LDSS + l16]      = (bf16)e0[r];
      myP[row * LDSS + 16 + l16] = (bf16)e1[r];
    }
    int ko = half * 8;
    const bf16* pp = &myP[l16 * LDSS + ko];
    v16bf ap = join16(*reinterpret_cast<const v8bf*>(pp),
                      *reinterpret_cast<const v8bf*>(pp + 16));

    // ---- P(16x32) . V(32x64): 4 WMMAs over dk chunks ----
#pragma unroll
    for (int n = 0; n < 4; ++n)
      oacc[n] = wmma_bf16(ap, bv[n], oacc[n]);

    // ---- stage next tile into the other buffer ----
    if (jt + 1 < NJ) storeKV((jt + 1) & 1);
    __syncthreads();
  }

  // ---- normalize and write head-merged fp32 context ----
  int b = bh >> 4, h = bh & (H_ - 1);
  float inv[8];
#pragma unroll
  for (int r = 0; r < 8; ++r) inv[r] = 1.0f / lrow[r];
#pragma unroll
  for (int n = 0; n < 4; ++n) {
#pragma unroll
    for (int r = 0; r < 8; ++r) {
      int s  = q0 + r + 8 * half;
      int dk = 16 * n + l16;
      ctx[((size_t)b * S_ + s) * D_ + h * DK_ + dk] = oacc[n][r] * inv[r];
    }
  }
}

// ---------------------------------------------------------------------------
// Launch: Q/K/V projections (bf16 head-split / transposed), flash attention,
// output projection. Workspace: Qh(16.8MB) Kh(16.8MB) Vt(16.8MB) ctx(33.5MB).
// ---------------------------------------------------------------------------
extern "C" void kernel_launch(void* const* d_in, const int* in_sizes, int n_in,
                              void* d_out, int out_size, void* d_ws, size_t ws_size,
                              hipStream_t stream) {
  const float* q  = (const float*)d_in[0];
  const float* k  = (const float*)d_in[1];
  const float* v  = (const float*)d_in[2];
  const float* wq = (const float*)d_in[3];
  const float* bq = (const float*)d_in[4];
  const float* wk = (const float*)d_in[5];
  const float* bk = (const float*)d_in[6];
  const float* wv = (const float*)d_in[7];
  const float* bv = (const float*)d_in[8];
  const float* wo = (const float*)d_in[9];
  const float* bo = (const float*)d_in[10];

  char* ws = (char*)d_ws;
  const size_t elems = (size_t)B_ * S_ * D_;       // 8388608
  bf16*  Qh  = (bf16*)(ws);
  bf16*  Kh  = (bf16*)(ws + elems * 2);
  bf16*  Vt  = (bf16*)(ws + elems * 4);
  float* ctx = (float*)(ws + elems * 6);

  dim3 ggrid(D_ / BN, MTOT / BM);                  // 8 x 64
  mha_gemm_kernel<0><<<ggrid, 256, 0, stream>>>(q, wq, bq, (void*)Qh);
  mha_gemm_kernel<0><<<ggrid, 256, 0, stream>>>(k, wk, bk, (void*)Kh);
  mha_gemm_kernel<1><<<ggrid, 256, 0, stream>>>(v, wv, bv, (void*)Vt);

  int ablocks = B_ * H_ * (S_ / 64);               // 2048
  mha_attn_kernel<<<ablocks, 128, 0, stream>>>(Qh, Kh, Vt, ctx);

  mha_gemm_kernel<2><<<ggrid, 256, 0, stream>>>(ctx, wo, bo, d_out);
}
